// GraphAttentionNetwork_40604620816582
// MI455X (gfx1250) — compile-verified, matrix operand order checked by reading, not run on hardware
//
#include <hip/hip_runtime.h>
#include <hip/hip_bf16.h>
#include <cstdint>

#define NNODES 4096
#define FIN    512
#define FHID   64
#define NHEAD  8
#define NCLASS 16
#define NIDX   512
#define ALPHA  0.2f
#define NEGBIG (-9e15f)
#define ADJW   (NNODES / 32)   // 128 words per row

typedef __attribute__((ext_vector_type(16))) __bf16 v16bf;
typedef __attribute__((ext_vector_type(8)))  float  v8f;

union VB16 { uint4 q[2]; unsigned short s[16]; v16bf v; };

__device__ __forceinline__ unsigned short f2b(float f) {
  unsigned int u = __builtin_bit_cast(unsigned int, f);
  u += 0x7FFFu + ((u >> 16) & 1u);          // round-to-nearest-even
  return (unsigned short)(u >> 16);
}
__device__ __forceinline__ float b2f(unsigned short b) {
  unsigned int u = ((unsigned int)b) << 16;
  return __builtin_bit_cast(float, u);
}
__device__ __forceinline__ uint4 pack8_bf16(const float* v) {
  union { unsigned short s[8]; uint4 q; } u;
#pragma unroll
  for (int i = 0; i < 8; ++i) u.s[i] = f2b(v[i]);
  return u.q;
}
// Build a WMMA 16-bit A/B operand: elems 0..7 at p, elems 8..15 at p+16 (bf16 units).
__device__ __forceinline__ v16bf load_ab(const unsigned short* p) {
  VB16 u;
  u.q[0] = *(const uint4*)(p);
  u.q[1] = *(const uint4*)(p + 16);
  return u.v;
}
__device__ __forceinline__ v8f wmma_bf16(v16bf a, v16bf b, v8f c) {
  return __builtin_amdgcn_wmma_f32_16x16x32_bf16(false, a, false, b, (short)0, c,
                                                 false, false);
}

// ---------------------------------------------------------------- adj -> bits
__global__ void k_pack_adj(const int* __restrict__ adj, uint32_t* __restrict__ bits,
                           int nwords) {
  int w = blockIdx.x * blockDim.x + threadIdx.x;
  if (w >= nwords) return;
  const int* p = adj + (size_t)w * 32;
  uint32_t m = 0;
#pragma unroll
  for (int b = 0; b < 32; ++b) m |= (p[b] > 0 ? 1u : 0u) << b;
  bits[w] = m;
}

// ----------------------------------------------------------- f32 -> bf16 copy
__global__ void k_cvt(const float* __restrict__ in, unsigned short* __restrict__ out,
                      int n) {
  int i = blockIdx.x * blockDim.x + threadIdx.x;
  if (i < n) out[i] = f2b(in[i]);
}

// ----------------------------------------- f32 [nmat][R][C] -> bf16 [nmat][C][R]
__global__ void k_cvt_t(const float* __restrict__ in, unsigned short* __restrict__ out,
                        int R, int C, int nmat) {
  int i = blockIdx.x * blockDim.x + threadIdx.x;
  int total = nmat * R * C;
  if (i >= total) return;
  int m = i / (R * C), rc = i % (R * C), r = rc / C, c = rc % C;
  out[(m * C + c) * R + r] = f2b(in[i]);
}

// -------------------------------------- hT[h][f][node] = (nodes @ W_h)^T, bf16
__global__ void __launch_bounds__(128)
k_gemm_h(const unsigned short* __restrict__ nodes_bf,   // [4096][512]
         const unsigned short* __restrict__ wt_bf,      // [H][64][512]
         unsigned short* __restrict__ hT) {             // [H][64][4096]
  const int lane = threadIdx.x & 31, wave = threadIdx.x >> 5;
  const int head = blockIdx.y;
  const int r0 = (blockIdx.x * 4 + wave) * 16;
  const int rsel = lane & 15;
  const int koff = (lane < 16) ? 0 : 8;
  v8f acc[4] = {};
  const unsigned short* arow = nodes_bf + (r0 + rsel) * FIN;
#pragma unroll                      // full unroll (16 iters): deep load pipelining
  for (int k = 0; k < FIN; k += 32) {
    v16bf a = load_ab(arow + k + koff);
#pragma unroll
    for (int c = 0; c < 4; ++c) {
      const unsigned short* brow = wt_bf + (head * FHID + c * 16 + rsel) * FIN;
      acc[c] = wmma_bf16(a, load_ab(brow + k + koff), acc[c]);
    }
  }
  const int rbase = r0 + ((lane < 16) ? 0 : 8);
#pragma unroll
  for (int c = 0; c < 4; ++c) {
    float t[8];
#pragma unroll
    for (int j = 0; j < 8; ++j) t[j] = acc[c][j];
    *(uint4*)(hT + (head * FHID + c * 16 + rsel) * NNODES + rbase) = pack8_bf16(t);
  }
}

// ------------------------------------------------------- src/dst projections L1
__global__ void k_srcdst(const unsigned short* __restrict__ hT,
                         const float* __restrict__ a_heads,   // [H][128]
                         float* __restrict__ src, float* __restrict__ dst) {
  int i = blockIdx.x * blockDim.x + threadIdx.x;   // h*4096+node
  if (i >= NHEAD * NNODES) return;
  int h = i >> 12, node = i & (NNODES - 1);
  float s = 0.f, d = 0.f;
  for (int n = 0; n < FHID; ++n) {
    float v = b2f(hT[(h * FHID + n) * NNODES + node]);
    s += v * a_heads[h * 2 * FHID + n];
    d += v * a_heads[h * 2 * FHID + FHID + n];
  }
  src[i] = s; dst[i] = d;
}

// ------------- fused masked-softmax attention + att@h + ELU + concat (layer 1)
__global__ void __launch_bounds__(128)
k_attn1(const uint32_t* __restrict__ bits,
        const unsigned short* __restrict__ hT,     // [H][64][4096]
        const float* __restrict__ src, const float* __restrict__ dst,  // [H][4096]
        unsigned short* __restrict__ x_bf) {       // [4096][512]
  const int lane = threadIdx.x & 31, wave = threadIdx.x >> 5;
  const int head = blockIdx.y;
  const int r0 = (blockIdx.x * 4 + wave) * 16;
  const int rsel = lane & 15;
  const int row = r0 + rsel;
  const int koff = (lane < 16) ? 0 : 8;
  const float srow = src[head * NNODES + row];
  const float* dsth = dst + head * NNODES;
  const uint32_t* brow = bits + row * ADJW;
  const unsigned short* hTb = hT + head * FHID * NNODES;

  float m = -INFINITY, ssum = 0.f;
  v8f acc[4] = {};
#pragma unroll 2
  for (int jb = 0; jb < NNODES; jb += 32) {
    // prefetch next chunk's B tiles + adjacency word (global_prefetch_b8)
    int jn = (jb + 32) & (NNODES - 1);
    __builtin_prefetch(brow + (jn >> 5), 0, 3);
#pragma unroll
    for (int c = 0; c < 4; ++c)
      __builtin_prefetch(hTb + (c * 16 + rsel) * NNODES + jn + koff, 0, 3);

    uint32_t aw = brow[jb >> 5];
    float dv[16];
    *(float4*)(dv + 0)  = *(const float4*)(dsth + jb + koff + 0);
    *(float4*)(dv + 4)  = *(const float4*)(dsth + jb + koff + 4);
    *(float4*)(dv + 8)  = *(const float4*)(dsth + jb + koff + 16);
    *(float4*)(dv + 12) = *(const float4*)(dsth + jb + koff + 20);
    float e[16], mloc = NEGBIG;
#pragma unroll
    for (int t = 0; t < 16; ++t) {
      int joff = (t & 7) + ((t >> 3) << 4) + koff;
      float z = srow + dv[t];
      z = (z > 0.f) ? z : ALPHA * z;
      e[t] = ((aw >> joff) & 1u) ? z : NEGBIG;
      mloc = fmaxf(mloc, e[t]);
    }
    mloc = fmaxf(mloc, __shfl_xor(mloc, 16, 32));
    float mnew = fmaxf(m, mloc);
    float scale = __expf(m - mnew);
    float p[16], psum = 0.f;
#pragma unroll
    for (int t = 0; t < 16; ++t) { p[t] = __expf(e[t] - mnew); psum += p[t]; }
    psum += __shfl_xor(psum, 16, 32);
    ssum = ssum * scale + psum;
    m = mnew;
#pragma unroll
    for (int j = 0; j < 8; ++j) {                 // rescale running accumulators
      float sc = __shfl(scale, (lane < 16) ? j : (8 + j), 32);
#pragma unroll
      for (int c = 0; c < 4; ++c) acc[c][j] *= sc;
    }
    VB16 ap;                                      // probs are already A-layout
#pragma unroll
    for (int t = 0; t < 16; ++t) ap.s[t] = f2b(p[t]);
#pragma unroll
    for (int c = 0; c < 4; ++c)
      acc[c] = wmma_bf16(ap.v, load_ab(hTb + (c * 16 + rsel) * NNODES + jb + koff),
                         acc[c]);
  }
  const int rbase = (lane < 16) ? 0 : 8;
#pragma unroll
  for (int j = 0; j < 8; ++j) {
    float sj = __shfl(ssum, (lane < 16) ? j : (8 + j), 32);
    float inv = 1.f / sj;
#pragma unroll
    for (int c = 0; c < 4; ++c) {
      float v = acc[c][j] * inv;
      v = (v > 0.f) ? v : (__expf(v) - 1.f);      // ELU fused with concat store
      x_bf[(r0 + rbase + j) * (NHEAD * FHID) + head * FHID + c * 16 + rsel] = f2b(v);
    }
  }
}

// ----------------------------------------- h2T[c][node] = (x @ W_out)^T, bf16
__global__ void __launch_bounds__(128)
k_gemm_h2(const unsigned short* __restrict__ x_bf,   // [4096][512]
          const unsigned short* __restrict__ woT,    // [16][512]
          unsigned short* __restrict__ h2T) {        // [16][4096]
  const int lane = threadIdx.x & 31, wave = threadIdx.x >> 5;
  const int r0 = (blockIdx.x * 4 + wave) * 16;
  const int rsel = lane & 15;
  const int koff = (lane < 16) ? 0 : 8;
  v8f acc = {};
  const unsigned short* arow = x_bf + (r0 + rsel) * (NHEAD * FHID);
  const unsigned short* brow = woT + rsel * (NHEAD * FHID);
#pragma unroll                      // full unroll (16 iters)
  for (int k = 0; k < NHEAD * FHID; k += 32)
    acc = wmma_bf16(load_ab(arow + k + koff), load_ab(brow + k + koff), acc);
  float t[8];
#pragma unroll
  for (int j = 0; j < 8; ++j) t[j] = acc[j];
  *(uint4*)(h2T + rsel * NNODES + r0 + ((lane < 16) ? 0 : 8)) = pack8_bf16(t);
}

__global__ void k_srcdst2(const unsigned short* __restrict__ h2T,
                          const float* __restrict__ a_out,
                          float* __restrict__ src2, float* __restrict__ dst2) {
  int i = blockIdx.x * blockDim.x + threadIdx.x;
  if (i >= NNODES) return;
  float s = 0.f, d = 0.f;
  for (int n = 0; n < NCLASS; ++n) {
    float v = b2f(h2T[n * NNODES + i]);
    s += v * a_out[n];
    d += v * a_out[NCLASS + n];
  }
  src2[i] = s; dst2[i] = d;
}

// --------------- layer-2 attention, only the idx-gathered 512 rows, f32 output
__global__ void __launch_bounds__(128)
k_attn2(const uint32_t* __restrict__ bits, const int* __restrict__ idx,
        const unsigned short* __restrict__ h2T,     // [16][4096]
        const float* __restrict__ src2, const float* __restrict__ dst2,
        float* __restrict__ out) {                  // [512][16]
  const int lane = threadIdx.x & 31, wave = threadIdx.x >> 5;
  const int t0 = (blockIdx.x * 4 + wave) * 16;
  const int rsel = lane & 15;
  const int row = idx[t0 + rsel];
  const int koff = (lane < 16) ? 0 : 8;
  const float srow = src2[row];
  const uint32_t* brow = bits + row * ADJW;
  float m = -INFINITY, ssum = 0.f;
  v8f acc = {};
#pragma unroll 2
  for (int jb = 0; jb < NNODES; jb += 32) {
    int jn = (jb + 32) & (NNODES - 1);
    __builtin_prefetch(brow + (jn >> 5), 0, 3);
    __builtin_prefetch(h2T + rsel * NNODES + jn + koff, 0, 3);

    uint32_t aw = brow[jb >> 5];
    float dv[16];
    *(float4*)(dv + 0)  = *(const float4*)(dst2 + jb + koff + 0);
    *(float4*)(dv + 4)  = *(const float4*)(dst2 + jb + koff + 4);
    *(float4*)(dv + 8)  = *(const float4*)(dst2 + jb + koff + 16);
    *(float4*)(dv + 12) = *(const float4*)(dst2 + jb + koff + 20);
    float e[16], mloc = NEGBIG;
#pragma unroll
    for (int t = 0; t < 16; ++t) {
      int joff = (t & 7) + ((t >> 3) << 4) + koff;
      float z = srow + dv[t];
      z = (z > 0.f) ? z : ALPHA * z;
      e[t] = ((aw >> joff) & 1u) ? z : NEGBIG;
      mloc = fmaxf(mloc, e[t]);
    }
    mloc = fmaxf(mloc, __shfl_xor(mloc, 16, 32));
    float mnew = fmaxf(m, mloc);
    float scale = __expf(m - mnew);
    float p[16], psum = 0.f;
#pragma unroll
    for (int t = 0; t < 16; ++t) { p[t] = __expf(e[t] - mnew); psum += p[t]; }
    psum += __shfl_xor(psum, 16, 32);
    ssum = ssum * scale + psum;
    m = mnew;
#pragma unroll
    for (int j = 0; j < 8; ++j) {
      float sc = __shfl(scale, (lane < 16) ? j : (8 + j), 32);
      acc[j] *= sc;
    }
    VB16 ap;
#pragma unroll
    for (int t = 0; t < 16; ++t) ap.s[t] = f2b(p[t]);
    acc = wmma_bf16(ap.v, load_ab(h2T + rsel * NNODES + jb + koff), acc);
  }
  const int rbase = (lane < 16) ? 0 : 8;
#pragma unroll
  for (int j = 0; j < 8; ++j) {
    float sj = __shfl(ssum, (lane < 16) ? j : (8 + j), 32);
    out[(t0 + rbase + j) * NCLASS + rsel] = acc[j] / sj;
  }
}

// ============================================================================
extern "C" void kernel_launch(void* const* d_in, const int* in_sizes, int n_in,
                              void* d_out, int out_size, void* d_ws, size_t ws_size,
                              hipStream_t stream) {
  (void)in_sizes; (void)n_in; (void)out_size; (void)ws_size;
  const float* nodes   = (const float*)d_in[0];
  const int*   adj     = (const int*)d_in[1];
  const int*   idx     = (const int*)d_in[2];
  const float* W_heads = (const float*)d_in[3];
  const float* a_heads = (const float*)d_in[4];
  const float* W_out   = (const float*)d_in[5];
  const float* a_out   = (const float*)d_in[6];
  float* out = (float*)d_out;

  char* ws = (char*)d_ws;
  size_t off = 0;
  auto carve = [&](size_t bytes) {
    void* p = ws + off;
    off += (bytes + 255) & ~(size_t)255;
    return p;
  };
  uint32_t*       adj_bits = (uint32_t*)      carve((size_t)NNODES * ADJW * 4);
  unsigned short* nodes_bf = (unsigned short*)carve((size_t)NNODES * FIN * 2);
  unsigned short* wt_bf    = (unsigned short*)carve((size_t)NHEAD * FHID * FIN * 2);
  unsigned short* hT       = (unsigned short*)carve((size_t)NHEAD * FHID * NNODES * 2);
  float*          src      = (float*)         carve((size_t)NHEAD * NNODES * 4);
  float*          dst      = (float*)         carve((size_t)NHEAD * NNODES * 4);
  unsigned short* x_bf     = (unsigned short*)carve((size_t)NNODES * NHEAD * FHID * 2);
  unsigned short* woT      = (unsigned short*)carve((size_t)NCLASS * NHEAD * FHID * 2);
  unsigned short* h2T      = (unsigned short*)carve((size_t)NCLASS * NNODES * 2);
  float*          src2     = (float*)         carve((size_t)NNODES * 4);
  float*          dst2     = (float*)         carve((size_t)NNODES * 4);

  const int B = 256;
  // 1) pack adjacency into a 2 MB bitmask (lives in L2 thereafter)
  int nwords = NNODES * ADJW;
  k_pack_adj<<<(nwords + B - 1) / B, B, 0, stream>>>(adj, adj_bits, nwords);
  // 2) precision conversions / transposes
  k_cvt<<<(NNODES * FIN + B - 1) / B, B, 0, stream>>>(nodes, nodes_bf, NNODES * FIN);
  k_cvt_t<<<(NHEAD * FIN * FHID + B - 1) / B, B, 0, stream>>>(W_heads, wt_bf, FIN,
                                                              FHID, NHEAD);
  k_cvt_t<<<(FIN * NCLASS + B - 1) / B, B, 0, stream>>>(W_out, woT, NHEAD * FHID,
                                                        NCLASS, 1);
  // 3) per-head feature GEMM (WMMA bf16), stored transposed
  k_gemm_h<<<dim3(NNODES / 64, NHEAD), 128, 0, stream>>>(nodes_bf, wt_bf, hT);
  // 4) attention logit projections
  k_srcdst<<<(NHEAD * NNODES + B - 1) / B, B, 0, stream>>>(hT, a_heads, src, dst);
  // 5) fused flash-softmax attention + att@h + ELU + head concat
  k_attn1<<<dim3(NNODES / 64, NHEAD), 128, 0, stream>>>(adj_bits, hT, src, dst, x_bf);
  // 6) output projection GEMM (WMMA bf16), stored transposed
  k_gemm_h2<<<NNODES / 64, 128, 0, stream>>>(x_bf, woT, h2T);
  k_srcdst2<<<(NNODES + B - 1) / B, B, 0, stream>>>(h2T, a_out, src2, dst2);
  // 7) layer-2 attention only for the 512 gathered rows -> final output
  k_attn2<<<NIDX / 64, 128, 0, stream>>>(adj_bits, idx, h2T, src2, dst2, out);
}
